// MambaAudioEncoder_9921374453882
// MI455X (gfx1250) — matrix operand order repaired and checked by reading, not compiled, not run on hardware
//
#include <hip/hip_runtime.h>
#include <hip/hip_bf16.h>

// ---------------------------------------------------------------------------
// Mamba audio encoder forward, gfx1250 (CDNA5, wave32, WMMA).
// Big GEMMs use v_wmma_f32_16x16x32_bf16 with guard-free 64x128 block tiles;
// B tiles staged transposed in LDS so fragments are contiguous b128 loads.
// A tiles staged via GLOBAL_LOAD_ASYNC_TO_LDS_B128 when available.
// ---------------------------------------------------------------------------

#define B_SZ     8
#define T_SZ     1024
#define IN_DIM   80
#define D_MODEL  512
#define N_LAYER  4
#define D_STATE  16
#define D_CONV   3
#define D_INNER  1024
#define DT_RANK  32
#define M_ROWS   (B_SZ * T_SZ)            // 8192
#define XZ_COLS  (2 * D_INNER)            // 2048
#define XDBL_COLS 64                      // DT_RANK + 2*D_STATE
#define XDBL_PAD  128                     // padded to a full BN tile

typedef __bf16 v16bf __attribute__((ext_vector_type(16)));
typedef float  v8f   __attribute__((ext_vector_type(8)));

union FragBF { v16bf v; unsigned short u[16]; uint4 q[2]; };
union U4S8   { uint4 q; unsigned short s[8]; };

// optional gfx1250 async global->LDS path (probe via __has_builtin)
#if defined(__gfx1250__) && __has_builtin(__builtin_amdgcn_global_load_async_to_lds_b128) && __has_builtin(__builtin_amdgcn_s_wait_asynccnt)
#define HAVE_ASYNC_LDS 1
typedef int v4i_async __attribute__((vector_size(4 * sizeof(int))));
typedef __attribute__((address_space(1))) v4i_async GlobalV4;
typedef __attribute__((address_space(3))) v4i_async LocalV4;
#else
#define HAVE_ASYNC_LDS 0
#endif

__device__ __forceinline__ unsigned short f2bf(float f) {
    union { float f; unsigned u; } v; v.f = f;
    unsigned r = v.u + 0x7FFFu + ((v.u >> 16) & 1u);   // round-to-nearest-even
    return (unsigned short)(r >> 16);
}
__device__ __forceinline__ float sigmf(float x) { return 1.0f / (1.0f + __expf(-x)); }
__device__ __forceinline__ float siluf(float x) { return x * sigmf(x); }
__device__ __forceinline__ float softplusf(float x) {
    return (x > 20.0f) ? x : log1pf(__expf(x));
}

// ---------------------------------------------------------------------------
// f32 -> bf16 elementwise conversion
// ---------------------------------------------------------------------------
__global__ __launch_bounds__(256) void cvt_bf16_kernel(
    const float* __restrict__ src, unsigned short* __restrict__ dst, int n) {
    int i = blockIdx.x * 256 + threadIdx.x;
    if (i < n) dst[i] = f2bf(src[i]);
}

// f32 (rows x 64) -> bf16 (rows x 128), zero-padded columns 64..127
__global__ __launch_bounds__(256) void cvt_pad_kernel(
    const float* __restrict__ src, unsigned short* __restrict__ dst, int n /*rows*128*/) {
    int i = blockIdx.x * 256 + threadIdx.x;
    if (i < n) {
        int r = i >> 7, c = i & 127;
        dst[i] = (c < XDBL_COLS) ? f2bf(src[r * XDBL_COLS + c]) : (unsigned short)0;
    }
}

// ---------------------------------------------------------------------------
// Input projection: h = x @ proj_w + proj_b   (K = 80, scalar fp32)
// ---------------------------------------------------------------------------
__global__ __launch_bounds__(256) void proj_kernel(
    const float* __restrict__ x, const float* __restrict__ pw,
    const float* __restrict__ pb, float* __restrict__ h) {
    int idx = blockIdx.x * 256 + threadIdx.x;         // M_ROWS * D_MODEL
    int row = idx >> 9;
    int c   = idx & (D_MODEL - 1);
    const float* xr = x + (size_t)row * IN_DIM;
    float acc = pb[c];
#pragma unroll 8
    for (int k = 0; k < IN_DIM; ++k) acc = fmaf(xr[k], pw[k * D_MODEL + c], acc);
    h[idx] = acc;
}

// ---------------------------------------------------------------------------
// LayerNorm over D_MODEL=512, one wave32 per row, bf16 output
// ---------------------------------------------------------------------------
__global__ __launch_bounds__(256) void ln_kernel(
    const float* __restrict__ h, const float* __restrict__ w,
    const float* __restrict__ b, unsigned short* __restrict__ out) {
    int wave = threadIdx.x >> 5, lane = threadIdx.x & 31;
    int row  = blockIdx.x * 8 + wave;
    const float* hr = h + (size_t)row * D_MODEL;
    float xv[16];
#pragma unroll
    for (int i = 0; i < 16; ++i) xv[i] = hr[lane + 32 * i];
    float s = 0.f;
#pragma unroll
    for (int i = 0; i < 16; ++i) s += xv[i];
#pragma unroll
    for (int m = 16; m >= 1; m >>= 1) s += __shfl_xor(s, m, 32);
    float mu = s * (1.0f / D_MODEL);
    float v = 0.f;
#pragma unroll
    for (int i = 0; i < 16; ++i) { float d = xv[i] - mu; v += d * d; }
#pragma unroll
    for (int m = 16; m >= 1; m >>= 1) v += __shfl_xor(v, m, 32);
    float rs = rsqrtf(v * (1.0f / D_MODEL) + 1e-5f);
    unsigned short* orow = out + (size_t)row * D_MODEL;
#pragma unroll
    for (int i = 0; i < 16; ++i) {
        int c = lane + 32 * i;
        orow[c] = f2bf((xv[i] - mu) * rs * w[c] + b[c]);
    }
}

// ---------------------------------------------------------------------------
// Guard-free tiled bf16 WMMA GEMM: Cout[M,N] = (Cin? Cin:0) + A[M,K]@B[K,N]
// Requires M%64==0, N%128==0, K%32==0.
// Block tile 64x128, 8 waves; wave tile 32x32 = 2x2 WMMA 16x16x32.
// A staged row-major, B staged TRANSPOSED so fragments are contiguous b128.
// LDS rows padded to 40 ushorts (80B): 16B-aligned + conflict-free frag reads.
// ---------------------------------------------------------------------------
#define BM 64
#define BN 128
#define BK 32
#define LPAD 40

__global__ __launch_bounds__(256) void gemm_bf16_wmma(
    const unsigned short* __restrict__ A,  // M x K (bf16 bits)
    const unsigned short* __restrict__ Bw, // K x N (bf16 bits)
    const float* __restrict__ Cin,         // may be null, M x N
    float* __restrict__ Cout,              // M x N
    int M, int N, int K) {
    __shared__ unsigned short As [BM][LPAD];   // [m][k]
    __shared__ unsigned short BsT[BN][LPAD];   // [n][k]  (transposed tile)

    const int tid  = threadIdx.x;
    const int lane = tid & 31;
    const int wv   = tid >> 5;     // 0..7
    const int wm   = wv >> 2;      // 0..1
    const int wn   = wv & 3;       // 0..3
    const int half = lane >> 4;    // 0..1
    const int l16  = lane & 15;

    const int blockM = blockIdx.y * BM;
    const int blockN = blockIdx.x * BN;

    v8f acc00 = {}, acc01 = {}, acc10 = {}, acc11 = {};

    const int aRow = tid >> 2;            // 0..63
    const int aCol = (tid & 3) << 3;      // 0,8,16,24

    for (int k0 = 0; k0 < K; k0 += BK) {
        __syncthreads();
        // ---- stage A tile (row-major, 16B per thread) ----
        {
            const unsigned short* gsrc = A + (size_t)(blockM + aRow) * K + k0 + aCol;
#if HAVE_ASYNC_LDS
            __builtin_amdgcn_global_load_async_to_lds_b128(
                (GlobalV4*)(void*)gsrc, (LocalV4*)&As[aRow][aCol], 0, 0);
#else
            *(uint4*)&As[aRow][aCol] = *(const uint4*)gsrc;
#endif
        }
        // ---- stage B tile transposed: BsT[n][k] = B[k0+k][blockN+n] ----
#pragma unroll
        for (int i = 0; i < 2; ++i) {
            int idx  = tid + (i << 8);        // 0..511
            int bRow = idx >> 4;              // 0..31  (k)
            int bCol = (idx & 15) << 3;       // 0..120 (n)
            U4S8 u;
            u.q = *(const uint4*)(Bw + (size_t)(k0 + bRow) * N + blockN + bCol);
#pragma unroll
            for (int e = 0; e < 8; ++e) BsT[bCol + e][bRow] = u.s[e];
        }
        if (k0 + BK < K) {
            __builtin_prefetch(A + (size_t)(blockM + aRow) * K + k0 + BK + aCol, 0, 1);
            __builtin_prefetch(Bw + (size_t)(k0 + BK + (tid >> 4)) * N + blockN, 0, 1);
        }
#if HAVE_ASYNC_LDS
        __builtin_amdgcn_s_wait_asynccnt(0);
#endif
        __syncthreads();

        // ---- fragments: contiguous 2 x b128 LDS loads each ----
        // A lane layout: u[0..7] = K half*8..+7, u[8..15] = K 16+half*8..+7
        FragBF a0, a1, b0, b1;
        {
            const int m0 = wm * 32 + l16;
            a0.q[0] = *(const uint4*)&As[m0][half * 8];
            a0.q[1] = *(const uint4*)&As[m0][16 + half * 8];
            a1.q[0] = *(const uint4*)&As[m0 + 16][half * 8];
            a1.q[1] = *(const uint4*)&As[m0 + 16][16 + half * 8];
        }
        // B lane layout: u[0..7] = K half*8..+7, u[8..15] = K 16+half*8..+7 at column n (BsT[n][])
        {
            const int n0 = wn * 32 + l16;
            b0.q[0] = *(const uint4*)&BsT[n0][half * 8];
            b0.q[1] = *(const uint4*)&BsT[n0][16 + half * 8];
            b1.q[0] = *(const uint4*)&BsT[n0 + 16][half * 8];
            b1.q[1] = *(const uint4*)&BsT[n0 + 16][16 + half * 8];
        }
        acc00 = __builtin_amdgcn_wmma_f32_16x16x32_bf16(false, a0.v, false, b0.v, (short)0, acc00, false, false);
        acc01 = __builtin_amdgcn_wmma_f32_16x16x32_bf16(false, a0.v, false, b1.v, (short)0, acc01, false, false);
        acc10 = __builtin_amdgcn_wmma_f32_16x16x32_bf16(false, a1.v, false, b0.v, (short)0, acc10, false, false);
        acc11 = __builtin_amdgcn_wmma_f32_16x16x32_bf16(false, a1.v, false, b1.v, (short)0, acc11, false, false);
    }

    // ---- epilogue: C/D layout (vgpr v -> m = v + 8*half, n = l16), unguarded ----
    if (Cin) {
#pragma unroll
        for (int v = 0; v < 8; ++v) {
            int m0 = blockM + wm * 32 + v + 8 * half;
            int m1 = m0 + 16;
            int n0 = blockN + wn * 32 + l16;
            int n1 = n0 + 16;
            Cout[(size_t)m0 * N + n0] = acc00[v] + Cin[(size_t)m0 * N + n0];
            Cout[(size_t)m0 * N + n1] = acc01[v] + Cin[(size_t)m0 * N + n1];
            Cout[(size_t)m1 * N + n0] = acc10[v] + Cin[(size_t)m1 * N + n0];
            Cout[(size_t)m1 * N + n1] = acc11[v] + Cin[(size_t)m1 * N + n1];
        }
    } else {
#pragma unroll
        for (int v = 0; v < 8; ++v) {
            int m0 = blockM + wm * 32 + v + 8 * half;
            int m1 = m0 + 16;
            int n0 = blockN + wn * 32 + l16;
            int n1 = n0 + 16;
            Cout[(size_t)m0 * N + n0] = acc00[v];
            Cout[(size_t)m0 * N + n1] = acc01[v];
            Cout[(size_t)m1 * N + n0] = acc10[v];
            Cout[(size_t)m1 * N + n1] = acc11[v];
        }
    }
}

// ---------------------------------------------------------------------------
// Causal depthwise conv (width 3) + SiLU over u = xz[..., :D_INNER]
// ---------------------------------------------------------------------------
__global__ __launch_bounds__(256) void conv_silu_kernel(
    const float* __restrict__ xz, const float* __restrict__ cw,
    const float* __restrict__ cb, float* __restrict__ uconv,
    unsigned short* __restrict__ uconv_bf) {
    int idx = blockIdx.x * 256 + threadIdx.x;     // B*T*D_INNER
    int d = idx & (D_INNER - 1);
    int t = (idx >> 10) & (T_SZ - 1);
    int b = idx >> 20;
    float acc = cb[d];
#pragma unroll
    for (int j = 0; j < D_CONV; ++j) {
        int ts = t - (D_CONV - 1) + j;
        if (ts >= 0)
            acc = fmaf(xz[(size_t)((b << 10) + ts) * XZ_COLS + d], cw[d * D_CONV + j], acc);
    }
    float s = siluf(acc);
    uconv[idx] = s;
    uconv_bf[idx] = f2bf(s);
}

// ---------------------------------------------------------------------------
// delta = softplus(dt @ dtproj_w + dtproj_b), dt = x_dbl[:, :32] (K=32 scalar)
// x_dbl rows padded to XDBL_PAD
// ---------------------------------------------------------------------------
__global__ __launch_bounds__(256) void delta_kernel(
    const float* __restrict__ xdbl, const float* __restrict__ dtw,
    const float* __restrict__ dtb, float* __restrict__ delta) {
    int idx = blockIdx.x * 256 + threadIdx.x;     // M_ROWS * D_INNER
    int row = idx >> 10;
    int d   = idx & (D_INNER - 1);
    const float* xr = xdbl + (size_t)row * XDBL_PAD;
    float acc = dtb[d];
#pragma unroll 8
    for (int r = 0; r < DT_RANK; ++r) acc = fmaf(xr[r], dtw[r * D_INNER + d], acc);
    delta[idx] = softplusf(acc);
}

// ---------------------------------------------------------------------------
// Selective scan: one thread per (b,d) channel, 16-wide state in registers,
// B/C staged through LDS 16 timesteps at a time. grid (D_INNER/256, B).
// ---------------------------------------------------------------------------
__global__ __launch_bounds__(256) void scan_kernel(
    const float* __restrict__ delta, const float* __restrict__ uconv,
    const float* __restrict__ xdbl, const float* __restrict__ A_log,
    float* __restrict__ y) {
    const int b = blockIdx.y;
    const int d = blockIdx.x * 256 + threadIdx.x;
    float Av[D_STATE];
#pragma unroll
    for (int s = 0; s < D_STATE; ++s) Av[s] = -__expf(A_log[d * D_STATE + s]);
    float h[D_STATE];
#pragma unroll
    for (int s = 0; s < D_STATE; ++s) h[s] = 0.f;

    __shared__ float BC[16][2 * D_STATE];   // cols 0..15 = B, 16..31 = C
    for (int chunk = 0; chunk < T_SZ; chunk += 16) {
        __syncthreads();
#pragma unroll
        for (int j = 0; j < 2; ++j) {
            int li = threadIdx.x + (j << 8);       // 0..511
            int r = li >> 5, c = li & 31;
            BC[r][c] = xdbl[(size_t)((b << 10) + chunk + r) * XDBL_PAD + DT_RANK + c];
        }
        __syncthreads();
        for (int i = 0; i < 16; ++i) {
            size_t rowi = (size_t)((b << 10) + chunk + i);
            float dl = delta[rowi * D_INNER + d];
            float ut = uconv[rowi * D_INNER + d];
            float du = dl * ut;
            float acc = 0.f;
#pragma unroll
            for (int s = 0; s < D_STATE; ++s) {
                h[s] = __expf(dl * Av[s]) * h[s] + du * BC[i][s];
                acc = fmaf(h[s], BC[i][D_STATE + s], acc);
            }
            y[rowi * D_INNER + d] = acc;
        }
    }
}

// ---------------------------------------------------------------------------
// Gating: ybf = bf16( (y_scan + D*u) * silu(z) ), z = xz[..., D_INNER:]
// ---------------------------------------------------------------------------
__global__ __launch_bounds__(256) void gate_kernel(
    const float* __restrict__ yscan, const float* __restrict__ uconv,
    const float* __restrict__ xz, const float* __restrict__ Dp,
    unsigned short* __restrict__ ybf) {
    int idx = blockIdx.x * 256 + threadIdx.x;     // M_ROWS * D_INNER
    int row = idx >> 10;
    int d   = idx & (D_INNER - 1);
    float z = xz[(size_t)row * XZ_COLS + D_INNER + d];
    float val = (yscan[idx] + Dp[d] * uconv[idx]) * siluf(z);
    ybf[idx] = f2bf(val);
}

// ---------------------------------------------------------------------------
// Masked mean pool over valid timesteps. grid (D_MODEL/256, B)
// ---------------------------------------------------------------------------
__global__ __launch_bounds__(256) void pool_kernel(
    const float* __restrict__ h, const int* __restrict__ lengths,
    float* __restrict__ pooled) {
    int c = blockIdx.x * 256 + threadIdx.x;
    int b = blockIdx.y;
    int len = lengths[b];
    float s = 0.f;
    for (int t = 0; t < len; ++t) s += h[(size_t)((b << 10) + t) * D_MODEL + c];
    float denom = (len < 1) ? 1.0f : (float)len;
    pooled[b * D_MODEL + c] = s / denom;
}

// ---------------------------------------------------------------------------
// Host-side orchestration
// ---------------------------------------------------------------------------
extern "C" void kernel_launch(void* const* d_in, const int* in_sizes, int n_in,
                              void* d_out, int out_size, void* d_ws, size_t ws_size,
                              hipStream_t stream) {
    (void)in_sizes; (void)n_in; (void)out_size; (void)ws_size;
    const float* x        = (const float*)d_in[0];
    const int*   lengths  = (const int*)d_in[1];
    const float* proj_w   = (const float*)d_in[2];
    const float* proj_b   = (const float*)d_in[3];
    const float* ln_w     = (const float*)d_in[4];
    const float* ln_b     = (const float*)d_in[5];
    const float* in_w     = (const float*)d_in[6];
    const float* conv_w   = (const float*)d_in[7];
    const float* conv_b   = (const float*)d_in[8];
    const float* xproj_w  = (const float*)d_in[9];
    const float* dtproj_w = (const float*)d_in[10];
    const float* dtproj_b = (const float*)d_in[11];
    const float* A_log    = (const float*)d_in[12];
    const float* Dp       = (const float*)d_in[13];
    const float* out_w    = (const float*)d_in[14];

    float* h      = (float*)d_out;                         // seq_out region, updated in place
    float* pooled = (float*)d_out + (size_t)M_ROWS * D_MODEL;

    // carve workspace
    char* w = (char*)d_ws;
    size_t off = 0;
    auto carve = [&](size_t bytes) -> void* {
        void* p = w + off;
        off = (off + bytes + 255) & ~(size_t)255;
        return p;
    };
    const size_t n_inw   = (size_t)N_LAYER * D_MODEL * XZ_COLS;
    const size_t n_xpw_p = (size_t)N_LAYER * D_INNER * XDBL_PAD;    // padded
    const size_t n_outw  = (size_t)N_LAYER * D_INNER * D_MODEL;
    unsigned short* inw_bf   = (unsigned short*)carve(n_inw * 2);
    unsigned short* xpw_bf   = (unsigned short*)carve(n_xpw_p * 2);
    unsigned short* outw_bf  = (unsigned short*)carve(n_outw * 2);
    unsigned short* hln_bf   = (unsigned short*)carve((size_t)M_ROWS * D_MODEL * 2);
    float*          xz       = (float*)carve((size_t)M_ROWS * XZ_COLS * 4);
    float*          uconv    = (float*)carve((size_t)M_ROWS * D_INNER * 4);
    unsigned short* uconv_bf = (unsigned short*)carve((size_t)M_ROWS * D_INNER * 2);
    float*          xdbl     = (float*)carve((size_t)M_ROWS * XDBL_PAD * 4);
    float*          delta    = (float*)carve((size_t)M_ROWS * D_INNER * 4);
    float*          yscan    = (float*)carve((size_t)M_ROWS * D_INNER * 4);
    unsigned short* ybf      = (unsigned short*)carve((size_t)M_ROWS * D_INNER * 2);

    // weight conversion (every call; deterministic / stateless)
    cvt_bf16_kernel<<<(int)((n_inw + 255) / 256), 256, 0, stream>>>(in_w, inw_bf, (int)n_inw);
    cvt_pad_kernel<<<(int)((n_xpw_p + 255) / 256), 256, 0, stream>>>(xproj_w, xpw_bf, (int)n_xpw_p);
    cvt_bf16_kernel<<<(int)((n_outw + 255) / 256), 256, 0, stream>>>(out_w, outw_bf, (int)n_outw);

    // input projection
    proj_kernel<<<(M_ROWS * D_MODEL) / 256, 256, 0, stream>>>(x, proj_w, proj_b, h);

    const int nElemInner = M_ROWS * D_INNER;
    for (int l = 0; l < N_LAYER; ++l) {
        ln_kernel<<<M_ROWS / 8, 256, 0, stream>>>(
            h, ln_w + l * D_MODEL, ln_b + l * D_MODEL, hln_bf);

        gemm_bf16_wmma<<<dim3(XZ_COLS / BN, M_ROWS / BM), 256, 0, stream>>>(
            hln_bf, inw_bf + (size_t)l * D_MODEL * XZ_COLS,
            nullptr, xz, M_ROWS, XZ_COLS, D_MODEL);

        conv_silu_kernel<<<nElemInner / 256, 256, 0, stream>>>(
            xz, conv_w + l * D_INNER * D_CONV, conv_b + l * D_INNER, uconv, uconv_bf);

        gemm_bf16_wmma<<<dim3(XDBL_PAD / BN, M_ROWS / BM), 256, 0, stream>>>(
            uconv_bf, xpw_bf + (size_t)l * D_INNER * XDBL_PAD,
            nullptr, xdbl, M_ROWS, XDBL_PAD, D_INNER);

        delta_kernel<<<nElemInner / 256, 256, 0, stream>>>(
            xdbl, dtproj_w + (size_t)l * DT_RANK * D_INNER, dtproj_b + l * D_INNER, delta);

        scan_kernel<<<dim3(D_INNER / 256, B_SZ), 256, 0, stream>>>(
            delta, uconv, xdbl, A_log + (size_t)l * D_INNER * D_STATE, yscan);

        gate_kernel<<<nElemInner / 256, 256, 0, stream>>>(
            yscan, uconv, xz, Dp + l * D_INNER, ybf);

        gemm_bf16_wmma<<<dim3(D_MODEL / BN, M_ROWS / BM), 256, 0, stream>>>(
            ybf, outw_bf + (size_t)l * D_INNER * D_MODEL,
            h, h, M_ROWS, D_MODEL, D_INNER);
    }

    pool_kernel<<<dim3(D_MODEL / 256, B_SZ), 256, 0, stream>>>(h, lengths, pooled);
}
